// CrossWindowAttention_69054484185181
// MI455X (gfx1250) — matrix operand order repaired and checked by reading, not compiled, not run on hardware
//
#include <hip/hip_runtime.h>
#include <hip/hip_bf16.h>

typedef __attribute__((ext_vector_type(16))) __bf16 v16bf;
typedef __attribute__((ext_vector_type(8)))  float  v8f;
typedef __attribute__((ext_vector_type(4)))  float  f32x4;

#define DIM   192
#define HEADS 6
#define HD    32
#define NTOK  64   // tokens per window
#define TPB   256

// ---------------- WMMA operand loaders (CDNA5 16x16x32 bf16 layouts) ----------------

// A-matrix 16x32 bf16: lane m = row (m = lane&15). For lanes 0-15: elems 0..7 = K[k0..k0+7],
// elems 8..15 = K[k0+16..k0+23]. For lanes 16-31: K[k0+8..15] then K[k0+24..31].
__device__ inline v16bf load_A(const __bf16* base, int stride, int row0, int k0, int lane) {
    int m  = lane & 15;
    int hi = (lane >> 4) & 1;
    const __bf16* p = base + (row0 + m) * stride + k0 + hi * 8;
    union { uint4 u[2]; v16bf v; } r;
    r.u[0] = *(const uint4*)(p);
    r.u[1] = *(const uint4*)(p + 16);
    return r.v;
}

// B-matrix 32x16 bf16 where memory holds B transposed: mem[n][k] (row n, contiguous k).
// lane n = col (lane&15); lanes 0-15 hold K[k0..k0+7],K[k0+16..k0+23]; lanes 16-31 hold K[k0+8..15],K[k0+24..31].
__device__ inline v16bf load_Bt(const __bf16* base, int stride, int n0, int k0, int lane) {
    int n  = lane & 15;
    int hi = (lane >> 4) & 1;
    const __bf16* p = base + (n0 + n) * stride + k0 + hi * 8;
    union { uint4 u[2]; v16bf v; } r;
    r.u[0] = *(const uint4*)(p);
    r.u[1] = *(const uint4*)(p + 16);
    return r.v;
}

#define WMMA_BF16(a, b, c) \
    __builtin_amdgcn_wmma_f32_16x16x32_bf16(false, (a), false, (b), (short)0, (c), false, false)

// ---------------- prep: transpose+convert weights to bf16, pre-gather bias ----------------

__global__ __launch_bounds__(TPB) void prep_kernel(
    const float* __restrict__ Wq, const float* __restrict__ Wkv, const float* __restrict__ Pw,
    const float* __restrict__ bias_table, const int* __restrict__ rel_index,
    __bf16* __restrict__ WqT, __bf16* __restrict__ WkvT, __bf16* __restrict__ PwT,
    float* __restrict__ rpb) {
    int i = blockIdx.x * blockDim.x + threadIdx.x;
    if (i < DIM * DIM) {               // WqT[n][k] = Wq[k][n]
        int n = i / DIM, k = i % DIM;
        WqT[i] = (__bf16)Wq[k * DIM + n];
        PwT[i] = (__bf16)Pw[k * DIM + n];
    }
    if (i < 2 * DIM * DIM) {           // WkvT[n][k] = Wkv[k][n], n in [0,384)
        int n = i / DIM, k = i % DIM;
        WkvT[i] = (__bf16)Wkv[k * (2 * DIM) + n];
    }
    if (i < HEADS * NTOK * NTOK) {     // rpb[h][r][c] = bias_table[rel_index[r][c]][h]
        int h = i / (NTOK * NTOK);
        int rc = i % (NTOK * NTOK);
        rpb[i] = bias_table[rel_index[rc] * HEADS + h];
    }
}

// ---------------- fused attention: one block per window ----------------

__global__ __launch_bounds__(TPB) void xwin_attn_kernel(
    const float* __restrict__ x, const float* __restrict__ y,
    const float* __restrict__ bq, const float* __restrict__ bkv,
    const float* __restrict__ proj_b,
    const __bf16* __restrict__ WqT, const __bf16* __restrict__ WkvT,
    const __bf16* __restrict__ PwT, const float* __restrict__ rpb,
    float* __restrict__ out) {

    // 48KB pool: phase A = x staging (24KB); head phase = sS[2][64][64] f32 (32KB)
    //                                                    + sP[2][64][64] bf16 (16KB)
    __shared__ alignas(16) char    pool[49152];
    __shared__ alignas(16) __bf16  sQ[NTOK][DIM];    // (Q + bq) * scale
    __shared__ alignas(16) __bf16  sK[NTOK][DIM];    // K + bias
    __shared__ alignas(16) __bf16  sVt[DIM][NTOK];   // V transposed: [d][token]
    __shared__ alignas(16) __bf16  sO[NTOK][DIM];    // y staging, then attention output

    __bf16* sXf = (__bf16*)pool;                 // x staging (64*192 bf16)
    float*  sS  = (float*)pool;                  // [2][64][64] exp/scores
    __bf16* sP  = (__bf16*)(pool + 32768);       // [2][64][64] probs
    __bf16* sYf = &sO[0][0];                     // y staging lives in sO

    const int tid  = threadIdx.x;
    const int wave = tid >> 5;
    const int lane = tid & 31;
    const int hi8  = ((lane >> 4) & 1) * 8;
    const int lcol = lane & 15;
    const size_t boff = (size_t)blockIdx.x * (NTOK * DIM);
    const float scale = 0.17677669529663687f;    // 32^-0.5

    // ---- stage x and y together (f32 -> bf16, non-temporal streaming reads) ----
    {
        const f32x4* xb = (const f32x4*)(x + boff);
        const f32x4* yb = (const f32x4*)(y + boff);
        for (int i = tid; i < (NTOK * DIM) / 4; i += TPB) {
            f32x4 fx = __builtin_nontemporal_load(xb + i);
            f32x4 fy = __builtin_nontemporal_load(yb + i);
            __bf16* dx = sXf + i * 4;
            __bf16* dy = sYf + i * 4;
            dx[0] = (__bf16)fx.x; dx[1] = (__bf16)fx.y; dx[2] = (__bf16)fx.z; dx[3] = (__bf16)fx.w;
            dy[0] = (__bf16)fy.x; dy[1] = (__bf16)fy.y; dy[2] = (__bf16)fy.z; dy[3] = (__bf16)fy.w;
        }
    }
    __syncthreads();

    // ---- Q = x @ Wq + bq (scaled): 2M x 3N register blocking per wave ----
    {
        const int mt0 = (wave & 1) * 2;
        const int nt0 = (wave >> 1) * 3;
        v8f acc[2][3] = {};
        #pragma unroll
        for (int kk = 0; kk < 6; ++kk) {
            v16bf bm[3], am[2];
            #pragma unroll
            for (int j = 0; j < 3; ++j) bm[j] = load_Bt(WqT, DIM, (nt0 + j) * 16, kk * 32, lane);
            #pragma unroll
            for (int i = 0; i < 2; ++i) am[i] = load_A(sXf, DIM, (mt0 + i) * 16, kk * 32, lane);
            #pragma unroll
            for (int i = 0; i < 2; ++i)
                #pragma unroll
                for (int j = 0; j < 3; ++j)
                    acc[i][j] = WMMA_BF16(am[i], bm[j], acc[i][j]);
        }
        #pragma unroll
        for (int i = 0; i < 2; ++i)
            #pragma unroll
            for (int j = 0; j < 3; ++j) {
                int col = (nt0 + j) * 16 + lcol;
                int rb  = (mt0 + i) * 16 + hi8;
                float bv = bq[col];
                #pragma unroll
                for (int r = 0; r < 8; ++r)
                    sQ[rb + r][col] = (__bf16)((acc[i][j][r] + bv) * scale);
            }
    }
    // no barrier: Q wrote sQ, KV writes sK/sVt (disjoint); both read staged data

    // ---- KV = y @ Wkv + bkv: 4M x 3N strips per wave (B reused 4x) ----
    {
        const int nt0 = wave * 3;
        v8f acc[4][3] = {};
        #pragma unroll
        for (int kk = 0; kk < 6; ++kk) {
            v16bf bm[3];
            #pragma unroll
            for (int j = 0; j < 3; ++j) bm[j] = load_Bt(WkvT, DIM, (nt0 + j) * 16, kk * 32, lane);
            #pragma unroll
            for (int i = 0; i < 4; ++i) {
                v16bf a = load_A(sYf, DIM, i * 16, kk * 32, lane);
                #pragma unroll
                for (int j = 0; j < 3; ++j)
                    acc[i][j] = WMMA_BF16(a, bm[j], acc[i][j]);
            }
        }
        #pragma unroll
        for (int i = 0; i < 4; ++i)
            #pragma unroll
            for (int j = 0; j < 3; ++j) {
                int col = (nt0 + j) * 16 + lcol;   // 0..383
                int rb  = i * 16 + hi8;
                float bv = bkv[col];
                if (col < DIM) {
                    #pragma unroll
                    for (int r = 0; r < 8; ++r) sK[rb + r][col] = (__bf16)(acc[i][j][r] + bv);
                } else {
                    int d = col - DIM;
                    #pragma unroll
                    for (int r = 0; r < 8; ++r) sVt[d][rb + r] = (__bf16)(acc[i][j][r] + bv);
                }
            }
    }
    __syncthreads();

    // ---- attention: 2 heads per iteration ----
    for (int it = 0; it < HEADS / 2; ++it) {
        const int h0 = it * 2;

        // S = Qh @ Kh^T + rpb : per wave, one N-strip x 2 M-tiles x 2 heads (4 WMMAs)
        {
            int nt  = wave >> 1;
            int mt0 = (wave & 1) * 2;
            #pragma unroll
            for (int hh = 0; hh < 2; ++hh) {
                int h = h0 + hh;
                v16bf b = load_Bt(&sK[0][0], DIM, nt * 16, h * HD, lane);
                const float* rp = rpb + (size_t)h * NTOK * NTOK;
                float* sSh = sS + hh * (NTOK * NTOK);
                #pragma unroll
                for (int i = 0; i < 2; ++i) {
                    v16bf a = load_A(&sQ[0][0], DIM, (mt0 + i) * 16, h * HD, lane);
                    v8f acc = {};
                    acc = WMMA_BF16(a, b, acc);
                    int col = nt * 16 + lcol;
                    int rb  = (mt0 + i) * 16 + hi8;
                    #pragma unroll
                    for (int r = 0; r < 8; ++r) {
                        int row = rb + r;
                        sSh[row * NTOK + col] = acc[r] + rp[row * NTOK + col];
                    }
                }
            }
        }
        __syncthreads();

        // softmax: 128 rows (2 heads), 2 threads per row, pair shuffle reductions
        {
            int row = tid >> 1;               // 0..127
            int hh  = row >> 6;
            int r   = row & 63;
            int j0  = (tid & 1) * 32;
            float*  srow = sS + hh * (NTOK * NTOK) + r * NTOK;
            __bf16* prow = sP + hh * (NTOK * NTOK) + r * NTOK;
            float mx = -1e30f;
            #pragma unroll
            for (int j = 0; j < 32; ++j) mx = fmaxf(mx, srow[j0 + j]);
            mx = fmaxf(mx, __shfl_xor(mx, 1, 2));
            float s = 0.f;
            #pragma unroll
            for (int j = 0; j < 32; ++j) {
                float e = __expf(srow[j0 + j] - mx);
                srow[j0 + j] = e;
                s += e;
            }
            s += __shfl_xor(s, 1, 2);
            float inv = 1.f / s;
            #pragma unroll
            for (int j = 0; j < 32; ++j)
                prow[j0 + j] = (__bf16)(srow[j0 + j] * inv);
        }
        __syncthreads();

        // Oh = P @ Vh : per wave one 16x16 tile per head (2 heads, K=64 -> 4 WMMAs)
        {
            int mt = wave & 3, nt = wave >> 2;
            #pragma unroll
            for (int hh = 0; hh < 2; ++hh) {
                int h = h0 + hh;
                const __bf16* sPh = sP + hh * (NTOK * NTOK);
                v8f acc = {};
                #pragma unroll
                for (int kk = 0; kk < 2; ++kk) {
                    v16bf a = load_A(sPh, NTOK, mt * 16, kk * 32, lane);
                    v16bf b = load_Bt(&sVt[0][0], NTOK, h * HD + nt * 16, kk * 32, lane);
                    acc = WMMA_BF16(a, b, acc);
                }
                int col = nt * 16 + lcol;
                int rb  = mt * 16 + hi8;
                #pragma unroll
                for (int r = 0; r < 8; ++r)
                    sO[rb + r][h * HD + col] = (__bf16)acc[r];
            }
        }
        __syncthreads();
    }

    // ---- out = O @ proj_w + proj_b : 2M x 3N register blocking, NT stores ----
    {
        float* ob = out + boff;
        const int mt0 = (wave & 1) * 2;
        const int nt0 = (wave >> 1) * 3;
        v8f acc[2][3] = {};
        #pragma unroll
        for (int kk = 0; kk < 6; ++kk) {
            v16bf bm[3], am[2];
            #pragma unroll
            for (int j = 0; j < 3; ++j) bm[j] = load_Bt(PwT, DIM, (nt0 + j) * 16, kk * 32, lane);
            #pragma unroll
            for (int i = 0; i < 2; ++i) am[i] = load_A(&sO[0][0], DIM, (mt0 + i) * 16, kk * 32, lane);
            #pragma unroll
            for (int i = 0; i < 2; ++i)
                #pragma unroll
                for (int j = 0; j < 3; ++j)
                    acc[i][j] = WMMA_BF16(am[i], bm[j], acc[i][j]);
        }
        #pragma unroll
        for (int i = 0; i < 2; ++i)
            #pragma unroll
            for (int j = 0; j < 3; ++j) {
                int col = (nt0 + j) * 16 + lcol;
                int rb  = (mt0 + i) * 16 + hi8;
                float pb = proj_b[col];
                #pragma unroll
                for (int r = 0; r < 8; ++r)
                    __builtin_nontemporal_store(acc[i][j][r] + pb,
                                                ob + (size_t)(rb + r) * DIM + col);
            }
    }
}

// ---------------- launch ----------------

extern "C" void kernel_launch(void* const* d_in, const int* in_sizes, int n_in,
                              void* d_out, int out_size, void* d_ws, size_t ws_size,
                              hipStream_t stream) {
    const float* x          = (const float*)d_in[0];
    const float* y          = (const float*)d_in[1];
    const float* Wq         = (const float*)d_in[2];
    const float* bq         = (const float*)d_in[3];
    const float* Wkv        = (const float*)d_in[4];
    const float* bkv        = (const float*)d_in[5];
    const float* bias_table = (const float*)d_in[6];
    const float* proj_w     = (const float*)d_in[7];
    const float* proj_b     = (const float*)d_in[8];
    const int*   rel_index  = (const int*)d_in[9];
    float* out = (float*)d_out;

    const int nwin = in_sizes[0] / (NTOK * DIM);   // 8192

    // workspace layout (bytes): WqT | WkvT | PwT | rpb
    char* ws = (char*)d_ws;
    __bf16* WqT  = (__bf16*)(ws);                               // 192*192*2 = 73728
    __bf16* WkvT = (__bf16*)(ws + 73728);                       // 192*384*2 = 147456
    __bf16* PwT  = (__bf16*)(ws + 73728 + 147456);              // 73728
    float*  rpb  = (float*)(ws + 73728 + 147456 + 73728);       // 6*64*64*4 = 98304

    {
        int total = 2 * DIM * DIM;                              // 73728 covers everything
        int blocks = (total + TPB - 1) / TPB;
        prep_kernel<<<blocks, TPB, 0, stream>>>(Wq, Wkv, proj_w, bias_table, rel_index,
                                                WqT, WkvT, PwT, rpb);
    }
    xwin_attn_kernel<<<nwin, TPB, 0, stream>>>(x, y, bq, bkv, proj_b,
                                               WqT, WkvT, PwT, rpb, out);
}
